// MACCSRuleParse_33663953666598
// MI455X (gfx1250) — matrix operand order repaired and checked by reading, not compiled, not run on hardware
//
#include <hip/hip_runtime.h>
#include <hip/hip_bf16.h>

// ---------------------------------------------------------------------------
// GATv2 x2 + BN + mean-pool + FFN for MI455X (gfx1250, wave32, WMMA).
// Dense layers: v_wmma_f32_16x16x32_f16 (f16 A/B, f32 accum), 2 col-tiles
// per wave, vectorized (b128) LDS staging + fragment loads, uniform
// fast-path staging for interior tiles (no predication, wide loads).
// Irregular edge stages: f32 global atomics against L2-resident node tensors.
// ---------------------------------------------------------------------------

typedef __attribute__((ext_vector_type(16))) _Float16 v16h;
typedef __attribute__((ext_vector_type(8)))  _Float16 v8h;
typedef __attribute__((ext_vector_type(8)))  float    v8f;

#define NEG_SLOPE 0.2f
#define BN_EPS    1e-5f
#define MAXF      200     // max feature width in edge kernels (layer2)
#define MAXBN     400     // max feature width in BN kernels

// ---------------------------------------------------------------- utilities

__global__ void fill_f32_kernel(float* __restrict__ p, float v, long long n) {
  long long i = (long long)blockIdx.x * blockDim.x + threadIdx.x;
  if (i < n) p[i] = v;
}

__device__ inline void atomicMaxFloat(float* addr, float val) {
  // Standard monotone-bit trick; addr must be initialized to -inf bits.
  if (val >= 0.0f) atomicMax((int*)addr, __float_as_int(val));
  else             atomicMin((unsigned int*)addr, __float_as_uint(val));
}

__device__ inline v16h cat16(v8h lo, v8h hi) {
  return __builtin_shufflevector(lo, hi, 0, 1, 2, 3, 4, 5, 6, 7,
                                 8, 9, 10, 11, 12, 13, 14, 15);
}

// ------------------------------------------------------------- WMMA GEMM
// C[M,F] = act(A[M,K] @ B[K,F] + bias), A/B row-major f32, f16 multiply,
// f32 accumulate. Block = 128 threads = 4 waves. Block tile: 64 rows x 32
// cols; each wave owns 16 rows x 32 cols (two 16x16 accumulators, one A
// fragment reused). K tiled by 32. Interior tiles use unconditional
// contiguous loads; boundary tiles use clamped loads * {0,1} mask (the
// multiply cannot be folded away, so no exec-mask predication is emitted).

__global__ void gemm_wmma_kernel(const float* __restrict__ A,
                                 const float* __restrict__ B,
                                 const float* __restrict__ bias,
                                 float* __restrict__ C,
                                 int M, int K, int F, int act) {
  __shared__ __align__(16) _Float16 As[64][32];   // [row][k]
  __shared__ __align__(16) _Float16 Bs[32][32];   // [col][k] (transposed)
  const int tid  = threadIdx.x;
  const int wave = tid >> 5;
  const int lane = tid & 31;
  const int rowBase = blockIdx.x * 64;
  const int colBase = blockIdx.y * 32;

  v8f acc0 = {0.f, 0.f, 0.f, 0.f, 0.f, 0.f, 0.f, 0.f};
  v8f acc1 = {0.f, 0.f, 0.f, 0.f, 0.f, 0.f, 0.f, 0.f};

  // staging coordinates
  const int ar     = tid >> 1;            // A: row handled by this thread
  const int akhalf = (tid & 1) << 4;      // A: k-half 0 or 16
  const int agr    = rowBase + ar;
  const long long arowOff = (long long)((agr < M) ? agr : (M - 1)) * K;
  const int bc  = tid & 31;               // B: column handled by this thread
  const int bkg = (tid >> 5) << 3;        // B: k-group 0,8,16,24
  const int bgc = colBase + bc;
  const int bgcSafe = (bgc < F) ? bgc : (F - 1);

  const bool aRowsFull = (rowBase + 64 <= M);   // uniform across block
  const bool bColsFull = (colBase + 32 <= F);   // uniform across block

  for (int k0 = 0; k0 < K; k0 += 32) {
    const bool kFull = (k0 + 32 <= K);          // uniform across block

    // ---- stage A tile (64x32): 16 contiguous k per thread, 2x b128 store
    v8h pa0, pa1;
    if (aRowsFull && kFull) {
      const float* ap = A + arowOff + k0 + akhalf;   // fully in bounds
#pragma unroll
      for (int j = 0; j < 8; ++j) pa0[j] = (_Float16)ap[j];
#pragma unroll
      for (int j = 0; j < 8; ++j) pa1[j] = (_Float16)ap[8 + j];
    } else {
      const float rowMask = (agr < M) ? 1.0f : 0.0f;
#pragma unroll
      for (int j = 0; j < 8; ++j) {
        int gk = k0 + akhalf + j;
        float msk = (gk < K) ? rowMask : 0.0f;
        pa0[j] = (_Float16)(A[arowOff + ((gk < K) ? gk : (K - 1))] * msk);
      }
#pragma unroll
      for (int j = 0; j < 8; ++j) {
        int gk = k0 + akhalf + 8 + j;
        float msk = (gk < K) ? rowMask : 0.0f;
        pa1[j] = (_Float16)(A[arowOff + ((gk < K) ? gk : (K - 1))] * msk);
      }
    }
    *(v8h*)&As[ar][akhalf]     = pa0;
    *(v8h*)&As[ar][akhalf + 8] = pa1;

    // ---- stage B tile (32k x 32col, transposed): 8 k per thread, b128 store
    v8h pb;
    if (bColsFull && kFull) {
      const float* bp = B + (long long)(k0 + bkg) * F + bgc;
#pragma unroll
      for (int j = 0; j < 8; ++j) pb[j] = (_Float16)bp[(long long)j * F];
    } else {
      const float colMask = (bgc < F) ? 1.0f : 0.0f;
#pragma unroll
      for (int j = 0; j < 8; ++j) {
        int gk = k0 + bkg + j;
        float msk = (gk < K) ? colMask : 0.0f;
        pb[j] = (_Float16)(B[(long long)((gk < K) ? gk : (K - 1)) * F + bgcSafe] * msk);
      }
    }
    *(v8h*)&Bs[bc][bkg] = pb;

    // prefetch next A chunk (global_prefetch_b8)
    if (k0 + 32 < K) {
      int gk = k0 + 32 + akhalf;
      if (agr < M && gk < K) __builtin_prefetch(&A[arowOff + gk], 0, 0);
    }
    __syncthreads();

    // ---- fragments (vectorized ds_load_b128)
    // A: lane<16 -> row=lane, K 0..7 & 16..23 ; lane>=16 -> K 8..15 & 24..31
    const v8h* arp = (const v8h*)&As[wave * 16 + (lane & 15)][0];
    const int ga = (lane < 16) ? 0 : 1;
    v16h a = cat16(arp[ga], arp[2 + ga]);
    // B: lane<16 -> col=lane, K 0..7 & 16..23 ; lane>=16 -> K 8..15 & 24..31
    const int gb = (lane < 16) ? 0 : 1;
    const v8h* bp0 = (const v8h*)&Bs[lane & 15][0];
    v16h b0 = cat16(bp0[gb], bp0[gb + 2]);
    const v8h* bp1 = (const v8h*)&Bs[16 + (lane & 15)][0];
    v16h b1 = cat16(bp1[gb], bp1[gb + 2]);

    acc0 = __builtin_amdgcn_wmma_f32_16x16x32_f16(false, a, false, b0,
                                                  (short)0, acc0, false, false);
    acc1 = __builtin_amdgcn_wmma_f32_16x16x32_f16(false, a, false, b1,
                                                  (short)0, acc1, false, false);
    __syncthreads();
  }

  // store: acc[r] -> row = rowBase + wave*16 + (lane<16 ? r : r+8)
  const int col0 = colBase + (lane & 15);
  const int col1 = col0 + 16;
#pragma unroll
  for (int r = 0; r < 8; ++r) {
    int row = rowBase + wave * 16 + ((lane < 16) ? r : r + 8);
    if (row < M) {
      if (col0 < F) {
        float v = acc0[r] + (bias ? bias[col0] : 0.0f);
        if (act == 1) v = fmaxf(v, 0.0f);
        C[(long long)row * F + col0] = v;
      }
      if (col1 < F) {
        float v = acc1[r] + (bias ? bias[col1] : 0.0f);
        if (act == 1) v = fmaxf(v, 0.0f);
        C[(long long)row * F + col1] = v;
      }
    }
  }
}

// ------------------------------------------------- degree / self-loop attr

__global__ void deg_loop_accum_kernel(const int* __restrict__ ei,
                                      const float* __restrict__ eattr,
                                      float* __restrict__ deg,
                                      float* __restrict__ lattr,
                                      int E) {
  long long i = (long long)blockIdx.x * blockDim.x + threadIdx.x;
  if (i >= (long long)E * 18) return;
  int e = (int)(i / 18), k = (int)(i % 18);
  int dst = ei[(long long)E + e];
  atomicAdd(&lattr[(long long)dst * 18 + k], eattr[i]);
  if (k == 0) atomicAdd(&deg[dst], 1.0f);
}

__global__ void loop_div_kernel(float* __restrict__ lattr,
                                const float* __restrict__ deg, int Nn) {
  long long i = (long long)blockIdx.x * blockDim.x + threadIdx.x;
  if (i >= (long long)Nn * 18) return;
  lattr[i] = lattr[i] / fmaxf(deg[i / 18], 1.0f);
}

// -------------------------------------------------------- edge attention
// Wave-per-edge: m_f = xl[src][f] + xr[dst][f] + sum_k ea[k]*We[k][f];
// s = sum_f leaky(m_f)*att[f];  atomic-max into smax[dst].

__global__ void edge_attn_logits_kernel(const int* __restrict__ ei,
                                        const float* __restrict__ eattr,
                                        const float* __restrict__ lattr,
                                        const float* __restrict__ xl,
                                        const float* __restrict__ xr,
                                        const float* __restrict__ We,
                                        const float* __restrict__ att,
                                        float* __restrict__ s,
                                        float* __restrict__ smax,
                                        int E, int ET, int F) {
  __shared__ float sWe[18 * MAXF];
  __shared__ float sAtt[MAXF];
  __shared__ float sEA[4][18];
  const int tid  = threadIdx.x;
  const int wave = tid >> 5;
  const int lane = tid & 31;
  const long long e = (long long)blockIdx.x * 4 + wave;

  for (int i = tid; i < 18 * F; i += blockDim.x) sWe[i] = We[i];
  for (int i = tid; i < F; i += blockDim.x) sAtt[i] = att[i];

  int src = 0, dst = 0;
  if (e < ET) {
    const float* ea;
    if (e < E) { src = ei[e]; dst = ei[(long long)E + e]; ea = eattr + e * 18; }
    else       { src = dst = (int)(e - E); ea = lattr + (e - E) * 18; }
    if (lane < 18) sEA[wave][lane] = ea[lane];
  }
  __syncthreads();
  if (e >= ET) return;

  float part = 0.0f;
  for (int f = lane; f < F; f += 32) {
    float m = xl[(long long)src * F + f] + xr[(long long)dst * F + f];
#pragma unroll
    for (int k = 0; k < 18; ++k) m += sEA[wave][k] * sWe[k * F + f];
    m = (m > 0.0f) ? m : m * NEG_SLOPE;
    part += m * sAtt[f];
  }
#pragma unroll
  for (int off = 16; off > 0; off >>= 1) part += __shfl_down(part, off, 32);
  if (lane == 0) {
    s[e] = part;
    atomicMaxFloat(&smax[dst], part);
  }
}

__global__ void edge_softmax_denom_kernel(const int* __restrict__ ei,
                                          const float* __restrict__ s,
                                          const float* __restrict__ smax,
                                          float* __restrict__ ex,
                                          float* __restrict__ denom,
                                          int E, int ET) {
  long long e = (long long)blockIdx.x * blockDim.x + threadIdx.x;
  if (e >= ET) return;
  int dst = (e < E) ? ei[(long long)E + e] : (int)(e - E);
  float v = __expf(s[e] - smax[dst]);
  ex[e] = v;
  atomicAdd(&denom[dst], v);
}

__global__ void edge_aggregate_kernel(const int* __restrict__ ei,
                                      const float* __restrict__ ex,
                                      const float* __restrict__ denom,
                                      const float* __restrict__ xl,
                                      float* __restrict__ out,
                                      int E, int ET, int F) {
  const int wave = threadIdx.x >> 5;
  const int lane = threadIdx.x & 31;
  const long long e = (long long)blockIdx.x * 4 + wave;
  if (e >= ET) return;
  int src, dst;
  if (e < E) { src = ei[e]; dst = ei[(long long)E + e]; }
  else       { src = dst = (int)(e - E); }
  const float alpha = ex[e] / fmaxf(denom[dst], 1e-16f);
  for (int f = lane; f < F; f += 32)
    atomicAdd(&out[(long long)dst * F + f], xl[(long long)src * F + f] * alpha);
}

// ------------------------------------------------------- bias + activation

__global__ void bias_act_kernel(float* __restrict__ h,
                                const float* __restrict__ bias,
                                long long n, int F, int act) {
  long long i = (long long)blockIdx.x * blockDim.x + threadIdx.x;
  if (i >= n) return;
  float v = h[i] + bias[i % F];
  if (act == 1) v = fmaxf(v, 0.0f);
  h[i] = v;
}

// --------------------------------------------------------------- BatchNorm

__global__ void bn_stats_kernel(const float* __restrict__ h,
                                float* __restrict__ gsum,
                                float* __restrict__ gsq,
                                long long M, int F) {
  __shared__ float lsum[MAXBN];
  __shared__ float lsq[MAXBN];
  for (int i = threadIdx.x; i < F; i += blockDim.x) { lsum[i] = 0.f; lsq[i] = 0.f; }
  __syncthreads();
  const long long total = M * F;
  for (long long i = (long long)blockIdx.x * blockDim.x + threadIdx.x;
       i < total; i += (long long)gridDim.x * blockDim.x) {
    float v = h[i];
    int f = (int)(i % F);
    atomicAdd(&lsum[f], v);
    atomicAdd(&lsq[f], v * v);
  }
  __syncthreads();
  for (int i = threadIdx.x; i < F; i += blockDim.x) {
    atomicAdd(&gsum[i], lsum[i]);
    atomicAdd(&gsq[i], lsq[i]);
  }
}

__global__ void bn_apply_kernel(float* __restrict__ h,
                                const float* __restrict__ gsum,
                                const float* __restrict__ gsq,
                                const float* __restrict__ g,
                                const float* __restrict__ b,
                                long long M, int F) {
  long long i = (long long)blockIdx.x * blockDim.x + threadIdx.x;
  if (i >= M * F) return;
  int f = (int)(i % F);
  float mu  = gsum[f] / (float)M;
  float var = gsq[f] / (float)M - mu * mu;   // biased, matches jnp.var
  h[i] = (h[i] - mu) * rsqrtf(var + BN_EPS) * g[f] + b[f];
}

// ------------------------------------------------------------------ pooling

__global__ void pool_cnt_kernel(const int* __restrict__ batch,
                                float* __restrict__ cnt, int Nn) {
  long long i = (long long)blockIdx.x * blockDim.x + threadIdx.x;
  if (i < Nn) atomicAdd(&cnt[batch[i]], 1.0f);
}

__global__ void pool_sum_kernel(const int* __restrict__ batch,
                                const float* __restrict__ h,
                                float* __restrict__ pooled, int Nn, int F) {
  long long i = (long long)blockIdx.x * blockDim.x + threadIdx.x;
  if (i >= (long long)Nn * F) return;
  int row = (int)(i / F), f = (int)(i % F);
  atomicAdd(&pooled[(long long)batch[row] * F + f], h[i]);
}

__global__ void pool_div_kernel(float* __restrict__ pooled,
                                const float* __restrict__ cnt, int Gn, int F) {
  long long i = (long long)blockIdx.x * blockDim.x + threadIdx.x;
  if (i >= (long long)Gn * F) return;
  pooled[i] = pooled[i] / fmaxf(cnt[i / F], 1.0f);
}

// ------------------------------------------------------------ host helpers

static inline void fill_f32(float* p, float v, long long n, hipStream_t st) {
  long long b = (n + 255) / 256;
  fill_f32_kernel<<<(unsigned)b, 256, 0, st>>>(p, v, n);
}

static inline void gemm(const float* A, const float* B, const float* bias,
                        float* C, int M, int K, int F, int act, hipStream_t st) {
  dim3 grid((M + 63) / 64, (F + 31) / 32);
  gemm_wmma_kernel<<<grid, 128, 0, st>>>(A, B, bias, C, M, K, F, act);
}

static inline void batchnorm(float* h, const float* g, const float* b,
                             long long M, int F, float* bnsum, float* bnsq,
                             hipStream_t st) {
  fill_f32(bnsum, 0.0f, F, st);
  fill_f32(bnsq,  0.0f, F, st);
  bn_stats_kernel<<<512, 256, 0, st>>>(h, bnsum, bnsq, M, F);
  long long n = M * F;
  bn_apply_kernel<<<(unsigned)((n + 255) / 256), 256, 0, st>>>(h, bnsum, bnsq, g, b, M, F);
}

static void gat_layer(const float* hin, int Fin, int F,
                      const float* Wl, const float* bl,
                      const float* Wr, const float* br,
                      const float* We, const float* att, const float* bias,
                      const int* ei, const float* eattr, const float* lattr,
                      float* xl, float* xr, float* sbuf, float* exbuf,
                      float* smax, float* denom, float* out,
                      int Nn, int E, int ET, hipStream_t st) {
  gemm(hin, Wl, bl, xl, Nn, Fin, F, 0, st);
  gemm(hin, Wr, br, xr, Nn, Fin, F, 0, st);
  fill_f32(smax, -__builtin_huge_valf(), Nn, st);
  fill_f32(denom, 0.0f, Nn, st);
  fill_f32(out, 0.0f, (long long)Nn * F, st);
  edge_attn_logits_kernel<<<(ET + 3) / 4, 128, 0, st>>>(
      ei, eattr, lattr, xl, xr, We, att, sbuf, smax, E, ET, F);
  edge_softmax_denom_kernel<<<(ET + 255) / 256, 256, 0, st>>>(
      ei, sbuf, smax, exbuf, denom, E, ET);
  edge_aggregate_kernel<<<(ET + 3) / 4, 128, 0, st>>>(
      ei, exbuf, denom, xl, out, E, ET, F);
  // out + bias, then ReLU (reference: bn(relu(gat + bias)))
  long long n = (long long)Nn * F;
  bias_act_kernel<<<(unsigned)((n + 255) / 256), 256, 0, st>>>(out, bias, n, F, 1);
}

// ------------------------------------------------------------------ launch

extern "C" void kernel_launch(void* const* d_in, const int* in_sizes, int n_in,
                              void* d_out, int out_size, void* d_ws, size_t ws_size,
                              hipStream_t stream) {
  // inputs per setup_inputs() dict order
  const float* x     = (const float*)d_in[0];   // [N,16]
  const int*   ei    = (const int*)d_in[1];     // [2,E] (src row, dst row)
  const float* eattr = (const float*)d_in[2];   // [E,18]
  const int*   batch = (const int*)d_in[3];     // [N]
  const float* Wl1 = (const float*)d_in[4],  *bl1 = (const float*)d_in[5];
  const float* Wr1 = (const float*)d_in[6],  *br1 = (const float*)d_in[7];
  const float* We1 = (const float*)d_in[8],  *att1 = (const float*)d_in[9];
  const float* bias1 = (const float*)d_in[10];
  const float* Wl2 = (const float*)d_in[11], *bl2 = (const float*)d_in[12];
  const float* Wr2 = (const float*)d_in[13], *br2 = (const float*)d_in[14];
  const float* We2 = (const float*)d_in[15], *att2 = (const float*)d_in[16];
  const float* bias2 = (const float*)d_in[17];
  const float* W3 = (const float*)d_in[18],  *b3 = (const float*)d_in[19];
  const float* bn1_g = (const float*)d_in[20], *bn1_b = (const float*)d_in[21];
  const float* bn2_g = (const float*)d_in[22], *bn2_b = (const float*)d_in[23];
  const float* bn3_g = (const float*)d_in[24], *bn3_b = (const float*)d_in[25];
  const float* bnp_g = (const float*)d_in[26], *bnp_b = (const float*)d_in[27];
  const float* bnf_g = (const float*)d_in[28], *bnf_b = (const float*)d_in[29];
  const float* Wf1 = (const float*)d_in[30], *bf1 = (const float*)d_in[31];
  const float* Wf2 = (const float*)d_in[32], *bf2 = (const float*)d_in[33];
  const float* Wf3 = (const float*)d_in[34], *bf3 = (const float*)d_in[35];

  const int Nn = in_sizes[0] / 16;
  const int E  = in_sizes[2] / 18;
  const int Gg = out_size / 100;
  const int ET = E + Nn;                        // edges + self loops

  // workspace layout (~125M floats ~ 500 MB)
  float* w = (float*)d_ws;
  float* deg    = w; w += Nn;
  float* lattr  = w; w += (long long)Nn * 18;
  float* xl     = w; w += (long long)Nn * 200;
  float* xr     = w; w += (long long)Nn * 200;
  float* hA     = w; w += (long long)Nn * 400;
  float* hB     = w; w += (long long)Nn * 400;
  float* sbuf   = w; w += ET;
  float* exbuf  = w; w += ET;
  float* smax   = w; w += Nn;
  float* denom  = w; w += Nn;
  float* bnsum  = w; w += MAXBN;
  float* bnsq   = w; w += MAXBN;
  float* pooled = w; w += (long long)Gg * 400;
  float* cnt    = w; w += Gg;
  float* f1     = w; w += (long long)Gg * 200;
  float* f2     = w; w += (long long)Gg * 100;
  (void)n_in; (void)ws_size;

  // self-loop edge_attr = mean of incoming edge_attr per node
  fill_f32(deg, 0.0f, Nn, stream);
  fill_f32(lattr, 0.0f, (long long)Nn * 18, stream);
  {
    long long n = (long long)E * 18;
    deg_loop_accum_kernel<<<(unsigned)((n + 255) / 256), 256, 0, stream>>>(ei, eattr, deg, lattr, E);
    long long m = (long long)Nn * 18;
    loop_div_kernel<<<(unsigned)((m + 255) / 256), 256, 0, stream>>>(lattr, deg, Nn);
  }

  // GATv2 layer 1: 16 -> 100, then bn1
  gat_layer(x, 16, 100, Wl1, bl1, Wr1, br1, We1, att1, bias1,
            ei, eattr, lattr, xl, xr, sbuf, exbuf, smax, denom, hA,
            Nn, E, ET, stream);
  batchnorm(hA, bn1_g, bn1_b, Nn, 100, bnsum, bnsq, stream);

  // GATv2 layer 2: 100 -> 200, then bn2
  gat_layer(hA, 100, 200, Wl2, bl2, Wr2, br2, We2, att2, bias2,
            ei, eattr, lattr, xl, xr, sbuf, exbuf, smax, denom, hB,
            Nn, E, ET, stream);
  batchnorm(hB, bn2_g, bn2_b, Nn, 200, bnsum, bnsq, stream);

  // Linear 200 -> 400 (+bias), bn3
  gemm(hB, W3, b3, hA, Nn, 200, 400, 0, stream);
  batchnorm(hA, bn3_g, bn3_b, Nn, 400, bnsum, bnsq, stream);

  // global mean pool + bnp
  fill_f32(pooled, 0.0f, (long long)Gg * 400, stream);
  fill_f32(cnt, 0.0f, Gg, stream);
  pool_cnt_kernel<<<(Nn + 255) / 256, 256, 0, stream>>>(batch, cnt, Nn);
  {
    long long n = (long long)Nn * 400;
    pool_sum_kernel<<<(unsigned)((n + 255) / 256), 256, 0, stream>>>(batch, hA, pooled, Nn, 400);
    long long m = (long long)Gg * 400;
    pool_div_kernel<<<(unsigned)((m + 255) / 256), 256, 0, stream>>>(pooled, cnt, Gg, 400);
  }
  batchnorm(pooled, bnp_g, bnp_b, Gg, 400, bnsum, bnsq, stream);

  // FFN: 400 -> 200 (relu) -> 100 (relu) -> bnf -> 100
  gemm(pooled, Wf1, bf1, f1, Gg, 400, 200, 1, stream);
  gemm(f1, Wf2, bf2, f2, Gg, 200, 100, 1, stream);
  batchnorm(f2, bnf_g, bnf_b, Gg, 100, bnsum, bnsq, stream);
  gemm(f2, Wf3, bf3, (float*)d_out, Gg, 100, 100, 0, stream);
}
